// SimpleSampleTokenizer_88957362635161
// MI455X (gfx1250) — matrix operand order repaired and checked by reading, not compiled, not run on hardware
//
#include <hip/hip_runtime.h>
#include <math.h>

// ---------------- problem constants ----------------
#define Bn   16
#define Dd   768
#define HWn  32
#define Nn   1024          // HW*HW
#define Zz   256
#define TOPK 512
#define MAPE (Bn * 512 * 512)   // 4,194,304 per map
#define DECE (Bn * Zz * Nn)     // 4,194,304

typedef __attribute__((ext_vector_type(16))) __bf16 v16bf;
typedef __attribute__((ext_vector_type(8)))  float  v8f;
typedef __attribute__((ext_vector_type(4)))  unsigned int u32x4;
typedef __attribute__((ext_vector_type(8)))  int i32x8;
typedef __attribute__((ext_vector_type(4)))  int i32x4;

union ATile { v16bf v; u32x4 q[2]; };

__device__ __forceinline__ unsigned short f2bf(float f) {
    unsigned int u = __float_as_uint(f);
    u += 0x7FFFu + ((u >> 16) & 1u);          // round-to-nearest-even
    return (unsigned short)(u >> 16);
}
__device__ __forceinline__ float bf2f(unsigned short h) {
    return __uint_as_float(((unsigned int)h) << 16);
}

// ---------------- TDM contiguous block copy: global -> LDS ----------------
// 1-row tensor of 8-byte elements (bytes must be a multiple of 8).
// Issue from ONE wave; follow with s_wait_tensorcnt(0) on that wave + barrier.
__device__ __forceinline__ void tdm_copy_to_lds(void* ldsDst, const void* gsrc,
                                                unsigned bytes) {
    unsigned elems = bytes >> 3;                              // 8B units
    unsigned lds_off = (unsigned)(unsigned long long)ldsDst;  // low32 = LDS offset
    unsigned long long ga = (unsigned long long)gsrc;
    u32x4 g0;
    g0[0] = 1u;                                               // count=1, user mode
    g0[1] = lds_off;                                          // lds_addr
    g0[2] = (unsigned)ga;                                     // global_addr[31:0]
    g0[3] = (unsigned)((ga >> 32) & 0x1FFFFFFu) | (2u << 30); // addr[56:32], type=2
    i32x8 g1;
    g1[0] = (int)(3u << 16);                        // data_size=3 (8B)
    g1[1] = (int)((elems & 0xFFFFu) << 16);         // tensor_dim0[15:0]
    g1[2] = (int)((elems >> 16) | (1u << 16));      // tensor_dim0[31:16], tensor_dim1=1
    g1[3] = (int)((elems & 0xFFFFu) << 16);         // tile_dim0
    g1[4] = 1;                                      // tile_dim1=1, tile_dim2=0
    g1[5] = (int)elems;                             // tensor_dim0_stride[31:0]
    g1[6] = 0;
    g1[7] = 0;
    i32x4 g2; g2[0] = 1; g2[1] = 1; g2[2] = 0; g2[3] = 0;   // tensor_dim2/3 = 1
    i32x4 g3; g3[0] = 0; g3[1] = 0x10000; g3[2] = 0; g3[3] = 0; // tensor_dim4 = 1
#if defined(__clang_major__) && (__clang_major__ >= 23)
    __builtin_amdgcn_tensor_load_to_lds(g0, g1, g2, g3, g1, 0);
#else
    __builtin_amdgcn_tensor_load_to_lds(g0, g1, g2, g3, 0);
#endif
}

// fragment loaders (A: row-in-lane layout, B: col-in-lane layout; K-contiguous src)
__device__ __forceinline__ void load_a_frag(ATile& a, const unsigned short* arow, int k0) {
    const u32x4* ap = (const u32x4*)(arow + k0);
    a.q[0] = ap[0];            // K {0..7}+hi*8
    a.q[1] = ap[2];            // K {16..23}+hi*8
}
__device__ __forceinline__ void load_b_frag(ATile& b, const unsigned short* brow, int k0) {
    const u32x4* bq = (const u32x4*)(brow + k0);
    b.q[0] = bq[0];            // K {0..7}+hi*8
    b.q[1] = bq[2];            // K {16..23}+hi*8
}

// ---------------- pack / convert kernels ----------------

// out[b][c][r] = bf16(in[b][r][c])   (tiled transpose, batched via blockIdx.z)
__global__ __launch_bounds__(256) void transpose_to_bf16(
    const float* __restrict__ in, unsigned short* __restrict__ out,
    int R, int C, long inBStride, long outBStride) {
    const float* ib = in + (size_t)blockIdx.z * (size_t)inBStride;
    unsigned short* ob = out + (size_t)blockIdx.z * (size_t)outBStride;
    __shared__ float tile[32][33];
    int c0 = blockIdx.x * 32, r0 = blockIdx.y * 32;
    int tx = threadIdx.x & 31, ty = threadIdx.x >> 5;   // 32 x 8
#pragma unroll
    for (int i = 0; i < 32; i += 8) {
        int r = r0 + ty + i, c = c0 + tx;
        tile[ty + i][tx] = (r < R && c < C) ? ib[(size_t)r * C + c] : 0.f;
    }
    __syncthreads();
#pragma unroll
    for (int i = 0; i < 32; i += 8) {
        int c = c0 + ty + i, r = r0 + tx;
        if (c < C && r < R) ob[(size_t)c * R + r] = f2bf(tile[tx][ty + i]);
    }
}

__global__ __launch_bounds__(256) void convert_bf16(
    const float* __restrict__ in, unsigned short* __restrict__ out, int n) {
    int i = blockIdx.x * 256 + threadIdx.x;
    if (i < n) out[i] = f2bf(in[i]);
}

// ---------------- per-token mean / rsqrt(var+eps) ----------------
__global__ __launch_bounds__(256) void meanvar_kernel(
    const unsigned short* __restrict__ xbf, float* __restrict__ mu,
    float* __restrict__ rsig) {
    int wid = threadIdx.x >> 5, lane = threadIdx.x & 31;
    int tok = blockIdx.x * 8 + wid;                  // 0 .. 16383
    const u32x4* row = (const u32x4*)(xbf + (size_t)tok * Dd);  // 96 u32x4
    float s = 0.f, ss = 0.f;
    for (int i = lane; i < 96; i += 32) {
        u32x4 q = row[i];
#pragma unroll
        for (int c = 0; c < 4; ++c) {
            unsigned int w = q[c];
            float f0 = __uint_as_float(w << 16);
            float f1 = __uint_as_float(w & 0xFFFF0000u);
            s += f0 + f1;
            ss += f0 * f0 + f1 * f1;
        }
    }
#pragma unroll
    for (int m = 16; m >= 1; m >>= 1) {
        s  += __shfl_xor(s,  m, 32);
        ss += __shfl_xor(ss, m, 32);
    }
    if (lane == 0) {
        float mean = s * (1.f / (float)Dd);
        float var  = fmaxf(ss * (1.f / (float)Dd) - mean * mean, 0.f);
        mu[tok]   = mean;
        rsig[tok] = rsqrtf(var + 1e-5f);
    }
}

// ---------------- GEMM1 fused: score = sigmoid(relu(x@w1+b1)@w2+b2) ---------
// WG = 8 waves; 16 tokens x full 768 cols. A staged in LDS via TDM.
// Dual accumulation chains (even/odd K-chunks) keep loads in flight under WMMA.
__global__ __launch_bounds__(256) void gemm1_score_kernel(
    const unsigned short* __restrict__ xbf, const unsigned short* __restrict__ w1T,
    const float* __restrict__ b1, const float* __restrict__ w2,
    const float* __restrict__ b2, float* __restrict__ score) {
    __shared__ u32x4 lA4[16 * Dd / 8];     // 16 x 768 bf16 = 24 KB
    __shared__ float lScore[16];
    unsigned short* lA = (unsigned short*)lA4;

    int tid = threadIdx.x;
    int blk = blockIdx.x;                  // B * N/16 = 1024
    int b  = blk >> 6;
    int nb = (blk & 63) << 4;

    if (tid < 32) {                        // wave 0 drives the Tensor Data Mover
        tdm_copy_to_lds(lA4, xbf + ((size_t)(b * Nn + nb)) * Dd, 16 * Dd * 2);
        __builtin_amdgcn_s_wait_tensorcnt(0);
    }
    if (tid < 16) lScore[tid] = 0.f;
    __syncthreads();

    int wid = tid >> 5, lane = tid & 31;
    int col = lane & 15, hi = lane >> 4;
    float acc[8];
#pragma unroll
    for (int j = 0; j < 8; ++j) acc[j] = 0.f;

    const unsigned short* arow = lA + col * Dd + hi * 8;
    for (int t = 0; t < 6; ++t) {
        int e0 = (wid * 6 + t) << 4;
        int e  = e0 + col;
        const unsigned short* brow = w1T + (size_t)e * Dd + hi * 8;
        __builtin_prefetch(w1T + (size_t)(e + 16) * Dd, 0, 1);  // global_prefetch_b8
        v8f c0 = {}, c1 = {};
#pragma unroll
        for (int kk = 0; kk < 24; kk += 2) {
            ATile a0, b0, a1, b1f;
            load_a_frag(a0, arow, kk << 5);
            load_b_frag(b0, brow, kk << 5);
            load_a_frag(a1, arow, (kk + 1) << 5);
            load_b_frag(b1f, brow, (kk + 1) << 5);
            c0 = __builtin_amdgcn_wmma_f32_16x16x32_bf16(
                    false, a0.v, false, b0.v, (short)0, c0, false, false);
            c1 = __builtin_amdgcn_wmma_f32_16x16x32_bf16(
                    false, a1.v, false, b1f.v, (short)0, c1, false, false);
        }
        float b1v = b1[e], w2v = w2[e];
#pragma unroll
        for (int j = 0; j < 8; ++j)
            acc[j] += fmaxf(c0[j] + c1[j] + b1v, 0.f) * w2v;
    }
#pragma unroll
    for (int m = 8; m >= 1; m >>= 1)
#pragma unroll
        for (int j = 0; j < 8; ++j) acc[j] += __shfl_xor(acc[j], m, 16);
    if (col == 0) {
        int rb = hi * 8;
#pragma unroll
        for (int j = 0; j < 8; ++j) atomicAdd(&lScore[rb + j], acc[j]);
    }
    __syncthreads();
    if (tid < 16) {
        float t = lScore[tid] + b2[0];
        score[b * Nn + nb + tid] = 1.f / (1.f + __expf(-t));
    }
}

// ---------------- global min/max of score ----------------
__global__ __launch_bounds__(256) void minmax_kernel(
    const float* __restrict__ score, float* __restrict__ mm) {
    __shared__ float slo[256], shi[256];
    float lo = 3.4e38f, hi = -3.4e38f;
    for (int i = threadIdx.x; i < Bn * Nn; i += 256) {
        float v = score[i];
        lo = fminf(lo, v); hi = fmaxf(hi, v);
    }
    slo[threadIdx.x] = lo; shi[threadIdx.x] = hi;
    __syncthreads();
    for (int s = 128; s > 0; s >>= 1) {
        if (threadIdx.x < s) {
            slo[threadIdx.x] = fminf(slo[threadIdx.x], slo[threadIdx.x + s]);
            shi[threadIdx.x] = fmaxf(shi[threadIdx.x], shi[threadIdx.x + s]);
        }
        __syncthreads();
    }
    if (threadIdx.x == 0) { mm[0] = slo[0]; mm[1] = shi[0]; }
}

// ---------------- top-512 set selection (rank by counting) ----------------
__global__ __launch_bounds__(1024) void select_kernel(
    const float* __restrict__ score, int* __restrict__ posmap,
    int* __restrict__ top_idx) {
    int b = blockIdx.x;
    __shared__ float s[Nn];
    __shared__ int counter;
    int t = threadIdx.x;
    if (t == 0) counter = 0;
    s[t] = score[b * Nn + t];
    __syncthreads();
    float my = s[t];
    int rank = 0;
    for (int j = 0; j < Nn; ++j) {
        float v = s[j];
        rank += (v > my) || (v == my && j < t);
    }
    int pos = -1;
    if (rank < TOPK) {
        pos = atomicAdd(&counter, 1);      // slot order irrelevant: scatter uses same map
        top_idx[b * TOPK + pos] = t;
    }
    posmap[b * Nn + t] = pos;
}

// ---------------- gather: layernorm * score -> bf16 rows ----------------
__global__ __launch_bounds__(256) void gather_kernel(
    const unsigned short* __restrict__ xbf, const int* __restrict__ top_idx,
    const float* __restrict__ score, const float* __restrict__ mu,
    const float* __restrict__ rsig, unsigned short* __restrict__ gathered) {
    int blk = blockIdx.x;                  // B * 512
    int b = blk >> 9;
    int n = top_idx[blk];
    int tok = b * Nn + n;
    float sc = score[tok], m = mu[tok], r = rsig[tok];
    float ka = r * sc, kc = -m * r * sc;   // (x-m)*r*sc = x*ka + kc
    const unsigned short* src = xbf + (size_t)tok * Dd;
    unsigned short* dst = gathered + (size_t)blk * Dd;
    for (int d = threadIdx.x; d < Dd; d += 256)
        dst[d] = f2bf(bf2f(src[d]) * ka + kc);
}

// ---------------- GEMM2: sample_h = gathered @ wp + bp ----------------
__global__ __launch_bounds__(256) void gemm2_kernel(
    const unsigned short* __restrict__ gathered, const unsigned short* __restrict__ wpT,
    const float* __restrict__ bp, float* __restrict__ sample_h) {
    __shared__ u32x4 lA4[16 * Dd / 8];     // 24 KB
    unsigned short* lA = (unsigned short*)lA4;
    int blk = blockIdx.x;                  // B * 512/16 = 512
    int b  = blk >> 5;
    int kb = (blk & 31) << 4;
    if (threadIdx.x < 32) {
        tdm_copy_to_lds(lA4, gathered + ((size_t)(b * TOPK + kb)) * Dd, 16 * Dd * 2);
        __builtin_amdgcn_s_wait_tensorcnt(0);
    }
    __syncthreads();
    int lane = threadIdx.x & 31, wid = threadIdx.x >> 5;
    int col = lane & 15, hi = lane >> 4;
    const unsigned short* arow = lA + col * Dd + hi * 8;
    for (int t = 0; t < 2; ++t) {
        int z0 = (wid * 2 + t) << 4;
        int z  = z0 + col;
        const unsigned short* brow = wpT + (size_t)z * Dd + hi * 8;
        v8f c0 = {}, c1 = {};
#pragma unroll
        for (int kk = 0; kk < 24; kk += 2) {
            ATile a0, b0, a1, b1f;
            load_a_frag(a0, arow, kk << 5);
            load_b_frag(b0, brow, kk << 5);
            load_a_frag(a1, arow, (kk + 1) << 5);
            load_b_frag(b1f, brow, (kk + 1) << 5);
            c0 = __builtin_amdgcn_wmma_f32_16x16x32_bf16(
                    false, a0.v, false, b0.v, (short)0, c0, false, false);
            c1 = __builtin_amdgcn_wmma_f32_16x16x32_bf16(
                    false, a1.v, false, b1f.v, (short)0, c1, false, false);
        }
        float bpv = bp[z];
#pragma unroll
        for (int j = 0; j < 8; ++j) {
            int row = kb + j + hi * 8;
            sample_h[((size_t)(b * TOPK + row)) * Zz + z] = c0[j] + c1[j] + bpv;
        }
    }
}

// ---------------- scatter into embT (B, N, Z) bf16 ----------------
__global__ __launch_bounds__(256) void scatter_emb_kernel(
    const float* __restrict__ sample_h, const int* __restrict__ posmap,
    const float* __restrict__ mask_token, unsigned short* __restrict__ embT) {
    int blk = blockIdx.x;                  // B * N
    int b = blk >> 10;
    int p = posmap[blk];
    int z = threadIdx.x;                   // 256 == Z
    float v = (p >= 0) ? sample_h[((size_t)(b * TOPK + p)) * Zz + z] : mask_token[z];
    embT[(size_t)blk * Zz + z] = f2bf(v);
}

// ---------------- GEMM3: dec = conv_w @ emb + conv_b ----------------
__global__ __launch_bounds__(256) void gemm3_kernel(
    const unsigned short* __restrict__ cw, const unsigned short* __restrict__ embT,
    const float* __restrict__ conv_b, float* __restrict__ dec) {
    __shared__ u32x4 lB4[16 * Zz / 8];     // 16 n-rows x 256 z = 8 KB
    unsigned short* lB = (unsigned short*)lB4;
    int blk = blockIdx.x;                  // B * N/16 = 1024
    int b  = blk >> 6;
    int n0 = (blk & 63) << 4;
    if (threadIdx.x < 32) {
        tdm_copy_to_lds(lB4, embT + ((size_t)(b * Nn + n0)) * Zz, 16 * Zz * 2);
        __builtin_amdgcn_s_wait_tensorcnt(0);
    }
    __syncthreads();
    int lane = threadIdx.x & 31, wid = threadIdx.x >> 5;
    int col = lane & 15, hi = lane >> 4;
    const unsigned short* brow = lB + col * Zz + hi * 8;
    for (int t = 0; t < 2; ++t) {
        int o0 = (wid * 2 + t) << 4;
        const unsigned short* arow = cw + (size_t)(o0 + col) * Zz + hi * 8;
        v8f c0 = {}, c1 = {};
#pragma unroll
        for (int kk = 0; kk < 8; kk += 2) {
            ATile a0, b0, a1, b1f;
            load_a_frag(a0, arow, kk << 5);
            load_b_frag(b0, brow, kk << 5);
            load_a_frag(a1, arow, (kk + 1) << 5);
            load_b_frag(b1f, brow, (kk + 1) << 5);
            c0 = __builtin_amdgcn_wmma_f32_16x16x32_bf16(
                    false, a0.v, false, b0.v, (short)0, c0, false, false);
            c1 = __builtin_amdgcn_wmma_f32_16x16x32_bf16(
                    false, a1.v, false, b1f.v, (short)0, c1, false, false);
        }
#pragma unroll
        for (int j = 0; j < 8; ++j) {
            int o = o0 + j + hi * 8;
            dec[((size_t)(b * Zz + o)) * Nn + (n0 + col)] = c0[j] + c1[j] + conv_b[o];
        }
    }
}

// ---------------- maps (binary + normalized score, 16x upsample) ------------
__global__ __launch_bounds__(256) void maps_kernel(
    const float* __restrict__ score, const int* __restrict__ posmap,
    const float* __restrict__ mm, float* __restrict__ bin_out,
    float* __restrict__ sc_out) {
    unsigned int idx = blockIdx.x * 256 + threadIdx.x;   // 0 .. 4194303
    int b = idx >> 18;
    int rem = idx & 262143;
    int i = rem >> 9, j = rem & 511;
    int n = (i >> 4) * HWn + (j >> 4);
    int t = b * Nn + n;
    bin_out[idx] = (posmap[t] >= 0) ? 1.f : 0.f;
    float smin = mm[0];
    float inv = 1.f / fmaxf(mm[1] - smin, 1e-5f);
    sc_out[idx] = (score[t] - smin) * inv;
}

// ---------------- host launcher ----------------
extern "C" void kernel_launch(void* const* d_in, const int* in_sizes, int n_in,
                              void* d_out, int out_size, void* d_ws, size_t ws_size,
                              hipStream_t stream) {
    const float* image      = (const float*)d_in[0];
    const float* w1         = (const float*)d_in[1];
    const float* b1         = (const float*)d_in[2];
    const float* w2         = (const float*)d_in[3];
    const float* b2         = (const float*)d_in[4];
    const float* wp         = (const float*)d_in[5];
    const float* bp         = (const float*)d_in[6];
    const float* mask_token = (const float*)d_in[7];
    const float* conv_w     = (const float*)d_in[8];
    const float* conv_b     = (const float*)d_in[9];
    float* out = (float*)d_out;

    char* wsp = (char*)d_ws;
    auto take = [&](size_t bytes) -> char* {
        char* p = wsp;
        wsp += (bytes + 255) & ~(size_t)255;
        return p;
    };
    unsigned short* xbf      = (unsigned short*)take((size_t)Bn * Nn * Dd * 2);
    unsigned short* w1T      = (unsigned short*)take((size_t)Dd * Dd * 2);
    unsigned short* wpT      = (unsigned short*)take((size_t)Zz * Dd * 2);
    unsigned short* cwbf     = (unsigned short*)take((size_t)Zz * Zz * 2);
    float*          mu       = (float*)take((size_t)Bn * Nn * 4);
    float*          rsig     = (float*)take((size_t)Bn * Nn * 4);
    float*          score    = (float*)take((size_t)Bn * Nn * 4);
    float*          mm       = (float*)take(256);
    int*            posmap   = (int*)take((size_t)Bn * Nn * 4);
    int*            top_idx  = (int*)take((size_t)Bn * TOPK * 4);
    unsigned short* gathered = (unsigned short*)take((size_t)Bn * TOPK * Dd * 2);
    float*          sample_h = (float*)take((size_t)Bn * TOPK * Zz * 4);
    unsigned short* embT     = (unsigned short*)take((size_t)Bn * Nn * Zz * 2);

    float* dec_out = out;
    float* bin_out = out + DECE;
    float* sc_out  = out + DECE + MAPE;

    // weight packing (bf16, K-contiguous layouts)
    transpose_to_bf16<<<dim3(24, 24, 1), 256, 0, stream>>>(w1, w1T, Dd, Dd, 0, 0);
    transpose_to_bf16<<<dim3(8, 24, 1), 256, 0, stream>>>(wp, wpT, Dd, Zz, 0, 0);
    convert_bf16<<<(Zz * Zz + 255) / 256, 256, 0, stream>>>(conv_w, cwbf, Zz * Zz);
    // x: (B,D,H,W) -> (B,N,D) bf16
    transpose_to_bf16<<<dim3(32, 24, Bn), 256, 0, stream>>>(
        image, xbf, Dd, Nn, (long)Dd * Nn, (long)Nn * Dd);

    meanvar_kernel<<<(Bn * Nn) / 8, 256, 0, stream>>>(xbf, mu, rsig);
    gemm1_score_kernel<<<Bn * (Nn / 16), 256, 0, stream>>>(xbf, w1T, b1, w2, b2, score);
    minmax_kernel<<<1, 256, 0, stream>>>(score, mm);
    select_kernel<<<Bn, 1024, 0, stream>>>(score, posmap, top_idx);
    gather_kernel<<<Bn * TOPK, 256, 0, stream>>>(xbf, top_idx, score, mu, rsig, gathered);
    gemm2_kernel<<<Bn * (TOPK / 16), 256, 0, stream>>>(gathered, wpT, bp, sample_h);
    scatter_emb_kernel<<<Bn * Nn, 256, 0, stream>>>(sample_h, posmap, mask_token, embT);
    gemm3_kernel<<<Bn * (Nn / 16), 256, 0, stream>>>(cwbf, embT, conv_b, dec_out);
    maps_kernel<<<MAPE / 256, 256, 0, stream>>>(score, posmap, mm, bin_out, sc_out);
}
